// HGRNAttention_13752485282224
// MI455X (gfx1250) — matrix operand order repaired and checked by reading, not compiled, not run on hardware
//
#include <hip/hip_runtime.h>
#include <hip/hip_bf16.h>
#include <cstdint>

typedef __attribute__((ext_vector_type(16))) __bf16 v16bf;
typedef __attribute__((ext_vector_type(8)))  __bf16 v8bf;
typedef __attribute__((ext_vector_type(4)))  __bf16 v4bf;
typedef __attribute__((ext_vector_type(8)))  float  v8f;
typedef __attribute__((ext_vector_type(4)))  float  v4f;

#define BM 128
#define BN 256
#define BK 32
#define LDSP 40      // padded LDS row stride (bf16 elems): 80B, avoids bank conflicts
#define SCAN_L 128   // scan chunk length; NC = S / SCAN_L = 32 chunks

// ---------------------------------------------------------------------------
// gfx1250 async global->LDS copy (16B per lane), tracked by ASYNCcnt.
// dst = 32-bit LDS byte address (low 32 bits of generic pointer), src = 64-bit VA.
// ---------------------------------------------------------------------------
__device__ __forceinline__ void async_ld_b128(uint32_t lds_addr, uint64_t gaddr) {
  asm volatile("global_load_async_to_lds_b128 %0, %1, off"
               :: "v"(lds_addr), "v"(gaddr)
               : "memory");
}
__device__ __forceinline__ void wait_async_le6() {
  asm volatile("s_wait_asynccnt 0x6" ::: "memory");
}
__device__ __forceinline__ void wait_async_le0() {
  asm volatile("s_wait_asynccnt 0x0" ::: "memory");
}

// ---------------------------------------------------------------------------
// fp32 -> bf16 conversion, 4-wide (n must be a multiple of 4 — all sizes are)
// ---------------------------------------------------------------------------
__global__ void convert_bf16_kernel(const float* __restrict__ src,
                                    __bf16* __restrict__ dst, size_t n4) {
  size_t idx = (size_t)blockIdx.x * blockDim.x + threadIdx.x;
  size_t stride = (size_t)gridDim.x * blockDim.x;
  const v4f* s = (const v4f*)src;
  v4bf* d = (v4bf*)dst;
  for (; idx < n4; idx += stride)
    d[idx] = __builtin_convertvector(s[idx], v4bf);
}

// ---------------------------------------------------------------------------
// C[M,N] (fp32) = A[M,K] (bf16) @ W[N,K]^T (bf16, nn.Linear layout)
// 256 threads = 8 waves; block tile 128x256; waves 2(M) x 4(N), 64x64 each.
// Double-buffered LDS fed by global_load_async_to_lds_b128 (ASYNCcnt),
// last iteration peeled so the steady-state loop is branch-free.
// ---------------------------------------------------------------------------
__global__ __launch_bounds__(256)
void gemm_bf16_kernel(const __bf16* __restrict__ A,
                      const __bf16* __restrict__ W,
                      float* __restrict__ C,
                      int M, int N, int K) {
  __shared__ __bf16 As[2][BM * LDSP];   // 2 x 10240 B
  __shared__ __bf16 Bs[2][BN * LDSP];   // 2 x 20480 B

  const int tid   = threadIdx.x;
  const int lane  = tid & 31;
  const int wave  = tid >> 5;      // 0..7
  const int waveM = wave >> 2;     // 0..1  (64-row slab)
  const int waveN = wave & 3;      // 0..3  (64-col slab)
  const int blockRow = blockIdx.y * BM;
  const int blockCol = blockIdx.x * BN;

  // WMMA lane decomposition (ISA 7.12.2)
  const int m  = lane & 15;
  const int kh = lane >> 4;

  v8f acc[4][4] = {};
  union Frag { v16bf v; v8bf h[2]; };

  // Stage one 128x32 A tile + 256x32 B tile into LDS buffer `sel`
  // (exactly 6 x 16B async ops per thread: 2 for A, 4 for B).
  auto stage = [&](int k0, int sel) {
#pragma unroll
    for (int it = 0; it < 2; ++it) {
      const int chunk = tid + 256 * it;        // 0..511
      const int row = chunk >> 2;
      const int col = (chunk & 3) * 8;
      const __bf16* g = A + (size_t)(blockRow + row) * K + k0 + col;
      async_ld_b128((uint32_t)(uintptr_t)&As[sel][row * LDSP + col],
                    (uint64_t)(uintptr_t)g);
    }
#pragma unroll
    for (int it = 0; it < 4; ++it) {
      const int chunk = tid + 256 * it;        // 0..1023
      const int row = chunk >> 2;
      const int col = (chunk & 3) * 8;
      const __bf16* g = W + (size_t)(blockCol + row) * K + k0 + col;
      async_ld_b128((uint32_t)(uintptr_t)&Bs[sel][row * LDSP + col],
                    (uint64_t)(uintptr_t)g);
    }
  };

  // 16 WMMAs on the tile currently resident in buffer `sel`.
  auto compute = [&](int sel) {
    Frag af[4];
#pragma unroll
    for (int i = 0; i < 4; ++i) {
      const __bf16* p = &As[sel][(waveM * 64 + i * 16 + m) * LDSP + kh * 8];
      af[i].h[0] = *(const v8bf*)p;
      af[i].h[1] = *(const v8bf*)(p + 16);
    }
#pragma unroll
    for (int j = 0; j < 4; ++j) {
      Frag bfr;
      const __bf16* p = &Bs[sel][(waveN * 64 + j * 16 + m) * LDSP + kh * 8];
      bfr.h[0] = *(const v8bf*)p;
      bfr.h[1] = *(const v8bf*)(p + 16);
#pragma unroll
      for (int i = 0; i < 4; ++i)
        acc[i][j] = __builtin_amdgcn_wmma_f32_16x16x32_bf16(
            false, af[i].v, false, bfr.v, (short)0, acc[i][j], false, false);
    }
  };

  stage(0, 0);
  int sel = 0;
  int k0 = 0;
  for (; k0 + BK < K; k0 += BK, sel ^= 1) {   // steady state: branch-free body
    stage(k0 + BK, sel ^ 1);  // 12 outstanding; oldest 6 = current tile
    wait_async_le6();         // async loads complete in order
    __syncthreads();          // tile k resident across all waves
    compute(sel);
    __syncthreads();          // all reads of buf[sel] done before next overwrite
  }
  wait_async_le0();           // peeled last tile
  __syncthreads();
  compute(sel);

  // ---- store C (streaming / non-temporal): vgpr v -> row v + 8*kh, col = lane&15
  const int cn     = lane & 15;
  const int rowOff = kh * 8;
#pragma unroll
  for (int i = 0; i < 4; ++i)
#pragma unroll
    for (int j = 0; j < 4; ++j) {
      const int cRow = blockRow + waveM * 64 + i * 16 + rowOff;
      const int cCol = blockCol + waveN * 64 + j * 16 + cn;
      float* cp = C + (size_t)cRow * N + cCol;
#pragma unroll
      for (int v = 0; v < 8; ++v)
        __builtin_nontemporal_store(acc[i][j][v], cp + (size_t)v * N);
    }
}

// ---------------------------------------------------------------------------
// Chunked scan, pass 1 (fused with gate math):
//   reads i (ibuf), f (fbuf); computes a = sigmoid(f), b = silu(i)*(1-a);
//   chunk-local scan h = a*h + b with h0 = 0; writes a -> fbuf, h_local -> ibuf;
//   emits per-chunk summaries (prod of a, final h). One thread per (b, c, d).
// ---------------------------------------------------------------------------
__global__ void scan_pass1_kernel(float* __restrict__ ibuf,
                                  float* __restrict__ fbuf,
                                  float* __restrict__ aprod,
                                  float* __restrict__ hend,
                                  int D, int NC) {
  const int idx = blockIdx.x * blockDim.x + threadIdx.x;   // ((b*NC + c)*D + d)
  const int d   = idx % D;
  const int bc  = idx / D;
  const int c   = bc % NC;
  const int b   = bc / NC;
  const size_t base = (((size_t)b * NC + c) * SCAN_L) * D + d;

  float h = 0.0f, p = 1.0f;
  for (int t = 0; t < SCAN_L; ++t) {
    const size_t o = base + (size_t)t * D;
    const float iv = ibuf[o];
    const float fv = fbuf[o];
    const float a  = 1.0f / (1.0f + __expf(-fv));
    const float bb = (iv / (1.0f + __expf(-iv))) * (1.0f - a);
    h = __builtin_fmaf(a, h, bb);
    p *= a;
    fbuf[o] = a;
    ibuf[o] = h;
  }
  aprod[idx] = p;
  hend[idx]  = h;
}

// ---------------------------------------------------------------------------
// Chunked scan, pass 2: per (b,d), sequentially combine NC chunk summaries.
// carry[c] = h entering chunk c.
// ---------------------------------------------------------------------------
__global__ void scan_pass2_kernel(const float* __restrict__ aprod,
                                  const float* __restrict__ hend,
                                  float* __restrict__ carry,
                                  int D, int NC, int total) {
  const int idx = blockIdx.x * blockDim.x + threadIdx.x;   // (b*D + d)
  if (idx >= total) return;
  const int d = idx % D;
  const int b = idx / D;
  float cy = 0.0f;
  for (int c = 0; c < NC; ++c) {
    const size_t o = ((size_t)b * NC + c) * D + d;
    carry[o] = cy;
    cy = __builtin_fmaf(aprod[o], cy, hend[o]);
  }
}

// ---------------------------------------------------------------------------
// Chunked scan, pass 3: h[t] = h_local[t] + carry * prefixProd(a[0..t]).
// ---------------------------------------------------------------------------
__global__ void scan_pass3_kernel(float* __restrict__ hbuf,
                                  const float* __restrict__ abuf,
                                  const float* __restrict__ carry,
                                  int D, int NC) {
  const int idx = blockIdx.x * blockDim.x + threadIdx.x;   // ((b*NC + c)*D + d)
  const float cy = carry[idx];
  if (cy == 0.0f) return;   // chunk 0 (and decayed chains) skip entirely
  const int d  = idx % D;
  const int bc = idx / D;
  const int c  = bc % NC;
  const int b  = bc / NC;
  const size_t base = (((size_t)b * NC + c) * SCAN_L) * D + d;

  float p = 1.0f;
  for (int t = 0; t < SCAN_L; ++t) {
    const size_t o = base + (size_t)t * D;
    p *= abuf[o];
    hbuf[o] = __builtin_fmaf(cy, p, hbuf[o]);
  }
}

// ---------------------------------------------------------------------------
// Gated RMSNorm: xg = h*silu(g); y = xg * rsqrt(mean(xg^2)+eps) * w  -> bf16
// ---------------------------------------------------------------------------
__global__ __launch_bounds__(256)
void rmsnorm_kernel(const float* __restrict__ h, const float* __restrict__ g,
                    const float* __restrict__ w, __bf16* __restrict__ y, int D) {
  __shared__ float red[256];
  const int row = blockIdx.x;
  const float* hp = h + (size_t)row * D;
  const float* gp = g + (size_t)row * D;
  __bf16*      yp = y + (size_t)row * D;

  float xg[8];
  float ss = 0.0f;
#pragma unroll
  for (int e = 0; e < 8; ++e) {
    int d = threadIdx.x + e * 256;
    float gv = gp[d];
    float v  = hp[d] * (gv / (1.0f + __expf(-gv)));
    xg[e] = v;
    ss += v * v;
  }
  red[threadIdx.x] = ss;
  __syncthreads();
  for (int s = 128; s > 0; s >>= 1) {
    if (threadIdx.x < s) red[threadIdx.x] += red[threadIdx.x + s];
    __syncthreads();
  }
  const float inv = rsqrtf(red[0] / (float)D + 1e-5f);
#pragma unroll
  for (int e = 0; e < 8; ++e) {
    int d = threadIdx.x + e * 256;
    yp[d] = (__bf16)(xg[e] * inv * w[d]);
  }
}

// ---------------------------------------------------------------------------
extern "C" void kernel_launch(void* const* d_in, const int* in_sizes, int n_in,
                              void* d_out, int out_size, void* d_ws, size_t ws_size,
                              hipStream_t stream) {
  (void)in_sizes; (void)n_in; (void)out_size; (void)ws_size;
  const int B = 4, S = 4096, D = 2048;
  const int M = B * S, N = D, K = D;
  const int NC = S / SCAN_L;              // 32 chunks

  const float* x  = (const float*)d_in[0];
  const float* Wi = (const float*)d_in[1];
  const float* Wf = (const float*)d_in[2];
  const float* Wg = (const float*)d_in[3];
  const float* wn = (const float*)d_in[4];
  const float* Wo = (const float*)d_in[5];
  float* out = (float*)d_out;

  char* ws = (char*)d_ws;
  size_t off = 0;
  auto alloc = [&](size_t bytes) -> void* {
    void* p = ws + off;
    off += (bytes + 255) & ~(size_t)255;
    return p;
  };
  __bf16* xb    = (__bf16*)alloc((size_t)M * K * sizeof(__bf16));
  __bf16* Wib   = (__bf16*)alloc((size_t)N * K * sizeof(__bf16));
  __bf16* Wfb   = (__bf16*)alloc((size_t)N * K * sizeof(__bf16));
  __bf16* Wgb   = (__bf16*)alloc((size_t)N * K * sizeof(__bf16));
  __bf16* Wob   = (__bf16*)alloc((size_t)N * K * sizeof(__bf16));
  float*  ibuf  = (float*)alloc((size_t)M * N * sizeof(float));  // i -> h
  float*  fbuf  = (float*)alloc((size_t)M * N * sizeof(float));  // f -> a
  float*  gbuf  = (float*)alloc((size_t)M * N * sizeof(float));  // g
  __bf16* yb    = (__bf16*)alloc((size_t)M * N * sizeof(__bf16));
  float*  aprod = (float*)alloc((size_t)B * NC * D * sizeof(float));
  float*  hend  = (float*)alloc((size_t)B * NC * D * sizeof(float));
  float*  carry = (float*)alloc((size_t)B * NC * D * sizeof(float));

  // 1) downconvert activations + weights to bf16 (4-wide)
  convert_bf16_kernel<<<4096, 256, 0, stream>>>(x,  xb,  (size_t)M * K / 4);
  convert_bf16_kernel<<<2048, 256, 0, stream>>>(Wi, Wib, (size_t)N * K / 4);
  convert_bf16_kernel<<<2048, 256, 0, stream>>>(Wf, Wfb, (size_t)N * K / 4);
  convert_bf16_kernel<<<2048, 256, 0, stream>>>(Wg, Wgb, (size_t)N * K / 4);
  convert_bf16_kernel<<<2048, 256, 0, stream>>>(Wo, Wob, (size_t)N * K / 4);

  // 2) three input projections (WMMA bf16, fp32 accum)
  dim3 gg(N / BN, M / BM);
  gemm_bf16_kernel<<<gg, 256, 0, stream>>>(xb, Wib, ibuf, M, N, K);
  gemm_bf16_kernel<<<gg, 256, 0, stream>>>(xb, Wfb, fbuf, M, N, K);
  gemm_bf16_kernel<<<gg, 256, 0, stream>>>(xb, Wgb, gbuf, M, N, K);

  // 3) fused gate + chunked linear recurrence along S
  const int chunkThreads = B * NC * D;            // 262144
  scan_pass1_kernel<<<chunkThreads / 256, 256, 0, stream>>>(ibuf, fbuf, aprod, hend, D, NC);
  scan_pass2_kernel<<<(B * D + 255) / 256, 256, 0, stream>>>(aprod, hend, carry, D, NC, B * D);
  scan_pass3_kernel<<<chunkThreads / 256, 256, 0, stream>>>(ibuf, fbuf, carry, D, NC);

  // 4) gated RMSNorm -> bf16 y
  rmsnorm_kernel<<<M, 256, 0, stream>>>(ibuf, gbuf, wn, yb, D);

  // 5) output projection -> fp32 d_out
  gemm_bf16_kernel<<<gg, 256, 0, stream>>>(yb, Wob, out, M, N, K);
}
